// DUPN_37409165148967
// MI455X (gfx1250) — matrix operand order repaired and checked by reading, not compiled, not run on hardware
//
#include <hip/hip_runtime.h>
#include <hip/hip_bf16.h>

typedef __attribute__((ext_vector_type(16))) __bf16        v16bf;
typedef __attribute__((ext_vector_type(4)))  __bf16        v4bf;
typedef __attribute__((ext_vector_type(8)))  float         v8f;
typedef __attribute__((ext_vector_type(8)))  unsigned int  u32x8;

#define T_   128
#define B_   256
#define D_   256
#define H_   512
#define NH   2048      // 4*H
#define KC   768       // D + H (fused input+recurrent K)
#define KLAB 4
#define KSTEPS (KC / 32)   // 24

__device__ __forceinline__ float sigmoidf_(float x) { return 1.0f / (1.0f + __expf(-x)); }

// ---------------- setup kernels ----------------

__global__ void k_convert_x(const float* __restrict__ x, __bf16* __restrict__ xb) {
    size_t i4 = (size_t)blockIdx.x * blockDim.x + threadIdx.x;
    if (i4 >= (size_t)T_ * B_ * D_ / 4) return;
    float4 f = ((const float4*)x)[i4];
    v4bf o = { (__bf16)f.x, (__bf16)f.y, (__bf16)f.z, (__bf16)f.w };
    ((v4bf*)xb)[i4] = o;
}

// WtT[n][k] (bf16, [NH][KC]) = k<D ? W_ih[n][k] : W_hh[n][k-D]
__global__ void k_pack_w(const float* __restrict__ W_ih, const float* __restrict__ W_hh,
                         __bf16* __restrict__ WtT) {
    int i4 = blockIdx.x * blockDim.x + threadIdx.x;
    if (i4 >= NH * KC / 4) return;
    int n = i4 / (KC / 4);
    int k = (i4 % (KC / 4)) * 4;
    const float* src = (k < D_) ? (W_ih + (size_t)n * D_ + k)
                                : (W_hh + (size_t)n * H_ + (k - D_));
    float4 f = *(const float4*)src;
    v4bf o = { (__bf16)f.x, (__bf16)f.y, (__bf16)f.z, (__bf16)f.w };
    *(v4bf*)(WtT + (size_t)n * KC + k) = o;
}

__global__ void k_init(float* __restrict__ c, __bf16* __restrict__ hb) {
    int i = blockIdx.x * blockDim.x + threadIdx.x;
    if (i < B_ * H_) { c[i] = 0.0f; hb[i] = (__bf16)0.0f; }
}

// w1 = v1 @ A1  (D floats),  w2 = v1 @ A2  (H floats)
__global__ void k_v1proj(const float* __restrict__ v1, const float* __restrict__ A1,
                         const float* __restrict__ A2, float* __restrict__ w1,
                         float* __restrict__ w2) {
    int j = blockIdx.x * blockDim.x + threadIdx.x;
    if (j < D_) {
        float s = 0.0f;
        for (int h = 0; h < H_; ++h) s += v1[h] * A1[(size_t)h * D_ + j];
        w1[j] = s;
    } else if (j < D_ + H_) {
        int jj = j - D_;
        float s = 0.0f;
        for (int h = 0; h < H_; ++h) s += v1[h] * A2[(size_t)h * H_ + jj];
        w2[jj] = s;
    }
}

// ---------------- per-step fused LSTM GEMM ----------------
// Z[B_, NH] = concat(x_t, h_{t-1}) [B_, KC] @ WtT^T  (WtT stored [NH][KC])
// 128 threads = 4 waves; block tile 64x64; wave = 16x64 strip (4 accum tiles).
// Double-buffered LDS with async global->LDS staging (ASYNCcnt pipeline).
__global__ __launch_bounds__(128)
void k_lstm_gemm(const __bf16* __restrict__ Xb, const __bf16* __restrict__ Hb,
                 const __bf16* __restrict__ WtT, float* __restrict__ Z, int t) {
    __shared__ __bf16 As[2][64][32];   // [buf][m][k], 64B rows
    __shared__ __bf16 Bs[2][64][32];   // [buf][n][k], 64B rows

    const __bf16* __restrict__ Xt = Xb + (size_t)t * B_ * D_;
    const int tid  = threadIdx.x;
    const int wave = tid >> 5;
    const int lane = tid & 31;
    const int hl   = lane & 15;
    const int hi   = lane >> 4;
    const int bM   = blockIdx.x * 64;
    const int bN   = blockIdx.y * 64;

    // each thread stages 2 A-chunks + 2 B-chunks (16B each) per K-slice
    const int r_ = tid >> 2;           // row 0..31 (+32 for second chunk)
    const int q_ = tid & 3;            // 16B chunk within 64B row

    auto stage = [&](int buf, int k0) {
        #pragma unroll
        for (int i = 0; i < 2; ++i) {
            int r = r_ + i * 32;
            int kk = k0 + q_ * 8;
            const __bf16* gsrc = (kk < D_) ? (Xt + (size_t)(bM + r) * D_ + kk)
                                           : (Hb + (size_t)(bM + r) * H_ + (kk - D_));
            unsigned lds_a = (unsigned)(size_t)&As[buf][r][q_ * 8];
            asm volatile("global_load_async_to_lds_b128 %0, %1, off"
                         :: "v"(lds_a), "v"(gsrc) : "memory");
        }
        #pragma unroll
        for (int i = 0; i < 2; ++i) {
            int r = r_ + i * 32;
            const __bf16* gsrc = WtT + (size_t)(bN + r) * KC + k0 + q_ * 8;
            unsigned lds_b = (unsigned)(size_t)&Bs[buf][r][q_ * 8];
            asm volatile("global_load_async_to_lds_b128 %0, %1, off"
                         :: "v"(lds_b), "v"(gsrc) : "memory");
        }
    };

    v8f acc[4] = {};

    stage(0, 0);                               // prologue: fill buffer 0

    for (int it = 0; it < KSTEPS; ++it) {
        const int cur = it & 1;
        if (it + 1 < KSTEPS) {
            stage(cur ^ 1, (it + 1) * 32);     // prefetch next K-slice
            // own group for iter `it` done (async loads complete in order);
            // the 4 ops just issued for `it+1` may still be in flight
            asm volatile("s_wait_asynccnt 0x4" ::: "memory");
        } else {
            asm volatile("s_wait_asynccnt 0x0" ::: "memory");
        }
        __syncthreads();

        // A fragment: lane (hl) holds row M = wave*16 + hl; hi selects K sub-halves
        u32x8 au;
        #pragma unroll
        for (int j = 0; j < 8; ++j) {
            int kb = 2 * j + ((j & 4) << 1) + hi * 8;   // {0,2,4,6,16,18,20,22} + hi*8
            au[j] = *(const unsigned int*)&As[cur][wave * 16 + hl][kb];
        }
        v16bf av = __builtin_bit_cast(v16bf, au);

        // load ALL B fragments first, then issue WMMAs back-to-back
        v16bf bv[4];
        #pragma unroll
        for (int s = 0; s < 4; ++s) {
            u32x8 bu;
            #pragma unroll
            for (int j = 0; j < 8; ++j) {
                int kb = 2 * j + ((j & 4) << 1) + hi * 8;   // {0,2,4,6,16,18,20,22} + hi*8
                bu[j] = *(const unsigned int*)&Bs[cur][s * 16 + hl][kb];
            }
            bv[s] = __builtin_bit_cast(v16bf, bu);
        }
        #pragma unroll
        for (int s = 0; s < 4; ++s) {
            acc[s] = __builtin_amdgcn_wmma_f32_16x16x32_bf16(
                false, av, false, bv[s], (short)0, acc[s], false, false);
        }
        __syncthreads();                       // all waves done reading `cur`
    }

    // C layout: VGPR r -> M = wave*16 + hi*8 + r, N = s*16 + hl
    #pragma unroll
    for (int s = 0; s < 4; ++s) {
        #pragma unroll
        for (int r = 0; r < 8; ++r) {
            int m = bM + wave * 16 + hi * 8 + r;
            int n = bN + s * 16 + hl;
            Z[(size_t)m * NH + n] = acc[s][r];
        }
    }
}

// ---------------- gate nonlinearity + state update ----------------
__global__ void k_lstm_gates(const float* __restrict__ Z, const float* __restrict__ b_ih,
                             const float* __restrict__ b_hh, float* __restrict__ c,
                             __bf16* __restrict__ hb, float* __restrict__ hs, int t) {
    int idx = blockIdx.x * blockDim.x + threadIdx.x;
    if (idx >= B_ * H_) return;
    int b = idx >> 9;          // /H_
    int n = idx & (H_ - 1);
    const float* zr = Z + (size_t)b * NH;
    float zi = zr[n]            + b_ih[n]            + b_hh[n];
    float zf = zr[H_ + n]       + b_ih[H_ + n]       + b_hh[H_ + n];
    float zg = zr[2 * H_ + n]   + b_ih[2 * H_ + n]   + b_hh[2 * H_ + n];
    float zo = zr[3 * H_ + n]   + b_ih[3 * H_ + n]   + b_hh[3 * H_ + n];
    float cp = (t == 0) ? 0.0f : c[idx];
    float cn = sigmoidf_(zf) * cp + sigmoidf_(zi) * tanhf(zg);
    float hn = sigmoidf_(zo) * tanhf(cn);
    c[idx]  = cn;
    hb[idx] = (__bf16)hn;
    hs[(size_t)t * B_ * H_ + idx] = hn;
}

// ---------------- attention scores: a[t,b] = sigmoid(w1.x + w2.h) ----------------
__global__ __launch_bounds__(256)
void k_scores(const float* __restrict__ x, const float* __restrict__ hs,
              const float* __restrict__ w1, const float* __restrict__ w2,
              float* __restrict__ a) {
    __shared__ float red[256];
    int tb = blockIdx.x;                 // t*B_ + b
    int t = tb / B_, b = tb % B_;
    const float* xr = x  + ((size_t)t * B_ + b) * D_;
    const float* hr = hs + ((size_t)t * B_ + b) * H_;
    float p = 0.0f;
    for (int i = threadIdx.x; i < KC; i += 256)
        p += (i < D_) ? w1[i] * xr[i] : w2[i - D_] * hr[i - D_];
    red[threadIdx.x] = p;
    __syncthreads();
    for (int s = 128; s > 0; s >>= 1) {
        if ((int)threadIdx.x < s) red[threadIdx.x] += red[threadIdx.x + s];
        __syncthreads();
    }
    if (threadIdx.x == 0) a[tb] = sigmoidf_(red[0]);
}

// ---------------- softmax-window attention pooling ----------------
__global__ __launch_bounds__(256)
void k_pool(const float* __restrict__ a, const float* __restrict__ hs,
            const int* __restrict__ lengths, float* __restrict__ out) {
    __shared__ float w[T_];
    __shared__ float red[256];
    int b = blockIdx.x >> 2;
    int k = blockIdx.x & 3;
    int tid = threadIdx.x;
    int len = lengths[b];
    int ts = max(len - KLAB, 0);
    int tk = ts + k;
    float* orow = out + ((size_t)b * KLAB + k) * H_;
    if (tk > len - 1) {                      // invalid slot -> exact zeros
        for (int h = tid; h < H_; h += 256) orow[h] = 0.0f;
        return;
    }
    float v = (tid <= tk && tid < T_) ? a[(size_t)tid * B_ + b] : -1e30f;
    red[tid] = v;  __syncthreads();
    for (int s = 128; s > 0; s >>= 1) {
        if (tid < s) red[tid] = fmaxf(red[tid], red[tid + s]);
        __syncthreads();
    }
    float m = red[0];  __syncthreads();
    float e = (tid <= tk && tid < T_) ? __expf(v - m) : 0.0f;
    red[tid] = e;  __syncthreads();
    for (int s = 128; s > 0; s >>= 1) {
        if (tid < s) red[tid] += red[tid + s];
        __syncthreads();
    }
    float inv = 1.0f / red[0];
    if (tid < T_) w[tid] = e * inv;
    __syncthreads();
    for (int h = tid; h < H_; h += 256) {
        float acc = 0.0f;
        for (int t = 0; t <= tk; ++t)
            acc += w[t] * hs[((size_t)t * B_ + b) * H_ + h];
        orow[h] = acc;
    }
}

// ---------------- launcher ----------------
extern "C" void kernel_launch(void* const* d_in, const int* in_sizes, int n_in,
                              void* d_out, int out_size, void* d_ws, size_t ws_size,
                              hipStream_t stream) {
    const float* x     = (const float*)d_in[0];
    const float* W_ih  = (const float*)d_in[1];
    const float* W_hh  = (const float*)d_in[2];
    const float* b_ih  = (const float*)d_in[3];
    const float* b_hh  = (const float*)d_in[4];
    const float* A1    = (const float*)d_in[5];
    const float* A2    = (const float*)d_in[6];
    const float* v1    = (const float*)d_in[7];
    const int*   lens  = (const int*)d_in[8];
    float* out = (float*)d_out;

    char* ws = (char*)d_ws;
    size_t off = 0;
    auto carve = [&](size_t bytes) -> char* {
        char* p = ws + off;
        off = (off + bytes + 255) & ~(size_t)255;
        return p;
    };
    __bf16* Xb  = (__bf16*)carve((size_t)T_ * B_ * D_ * 2);   // 16 MB
    __bf16* WtT = (__bf16*)carve((size_t)NH * KC * 2);        //  3 MB
    float*  Z   = (float*) carve((size_t)B_ * NH * 4);        //  2 MB
    float*  c   = (float*) carve((size_t)B_ * H_ * 4);        // .5 MB
    __bf16* hb  = (__bf16*)carve((size_t)B_ * H_ * 2);        // .25 MB
    float*  hs  = (float*) carve((size_t)T_ * B_ * H_ * 4);   // 64 MB
    float*  a   = (float*) carve((size_t)T_ * B_ * 4);        // 128 KB
    float*  w1  = (float*) carve((size_t)D_ * 4);
    float*  w2  = (float*) carve((size_t)H_ * 4);

    k_convert_x<<<(T_ * B_ * D_ / 4) / 256, 256, 0, stream>>>(x, Xb);
    k_pack_w<<<(NH * KC / 4) / 256, 256, 0, stream>>>(W_ih, W_hh, WtT);
    k_init<<<(B_ * H_) / 256, 256, 0, stream>>>(c, hb);
    k_v1proj<<<3, 256, 0, stream>>>(v1, A1, A2, w1, w2);

    dim3 ggrid(B_ / 64, NH / 64);   // 4 x 32
    for (int t = 0; t < T_; ++t) {
        k_lstm_gemm<<<ggrid, 128, 0, stream>>>(Xb, hb, WtT, Z, t);
        k_lstm_gates<<<(B_ * H_) / 256, 256, 0, stream>>>(Z, b_ih, b_hh, c, hb, hs, t);
    }

    k_scores<<<T_ * B_, 256, 0, stream>>>(x, hs, w1, w2, a);
    k_pool<<<B_ * KLAB, 256, 0, stream>>>(a, hs, lens, out);
}
